// SelfAttention_6476810682699
// MI455X (gfx1250) — matrix operand order repaired and checked by reading, not compiled, not run on hardware
//
#include <hip/hip_runtime.h>

// ---------------------------------------------------------------------------
// Causal multi-head self-attention for B=4, T=2048, C=1024, H=16, D=64.
// bf16 WMMA (f32 accumulate); flash-attention inner loop; consume-then-refill
// pipelined GEMMs (bounded register pressure, no fragment copies).
// Workspace usage: 88 MiB (xb 16 + weights 8 + k/q/vT/attn-out 4x16).
// ---------------------------------------------------------------------------

typedef __bf16 bf16_t;
typedef __attribute__((ext_vector_type(16))) __bf16 v16bf;
typedef __attribute__((ext_vector_type(8)))  float   v8f;

#define TSEQ 2048
#define CDIM 1024
#define NHEAD 16
#define HDIM 64

union Frag32B { v16bf v; uint4 q[2]; };

// A-matrix 16x32 bf16 fragment (ISA 7.12.2): lane<16 -> M=lane, K={0..7,16..23};
// lane>=16 -> M=lane-16, K={8..15,24..31}. Source row-major, row stride ld.
__device__ __forceinline__ v16bf load_frag_A(const bf16_t* base, int ld, int lane) {
  const int m  = lane & 15;
  const int hb = (lane >> 4) * 8;          // 0 or 8
  const bf16_t* p = base + m * ld + hb;
  Frag32B f;
  f.q[0] = *(const uint4*)(p);             // K = hb .. hb+7
  f.q[1] = *(const uint4*)(p + 16);        // K = 16+hb .. 16+hb+7
  return f.v;
}

// B-matrix 32x16 bf16 fragment: lane<16 -> N=lane, K={0..7,16..23}; lane>=16 -> N=lane-16,
// K={8..15,24..31}. Source addressed so element (k,n) = base + n*ld + k (i.e. B columns
// are contiguous rows of the source matrix).
__device__ __forceinline__ v16bf load_frag_B(const bf16_t* base, int ld, int lane) {
  const int n  = lane & 15;
  const int kk = (lane >> 4) * 8;          // 0 or 8
  const bf16_t* p = base + n * ld + kk;
  Frag32B f;
  f.q[0] = *(const uint4*)(p);
  f.q[1] = *(const uint4*)(p + 16);
  return f.v;
}

__device__ __forceinline__ v8f wmma_bf16(v16bf a, v16bf b, v8f c) {
  return __builtin_amdgcn_wmma_f32_16x16x32_bf16(false, a, false, b,
                                                 (short)0, c, false, false);
}

// ---------------------------------------------------------------------------
// fp32 -> bf16 conversion, 4 elements per thread per trip
// ---------------------------------------------------------------------------
__global__ void cvt_f32_bf16_v4(const float4* __restrict__ in,
                                ushort4* __restrict__ out, int n4) {
  int i = blockIdx.x * blockDim.x + threadIdx.x;
  const int stride = gridDim.x * blockDim.x;
  for (; i < n4; i += stride) {
    const float4 f = in[i];
    union { ushort4 u; bf16_t b[4]; } o;
    o.b[0] = (bf16_t)f.x; o.b[1] = (bf16_t)f.y;
    o.b[2] = (bf16_t)f.z; o.b[3] = (bf16_t)f.w;
    out[i] = o.u;
  }
}

// ---------------------------------------------------------------------------
// Pipelined wave-level GEMM body: 32x64 tile per wave.
//  - B fragments: consume-then-refill in place (single live range, no copies;
//    WAR dep bounds in-flight loads, refill covered by 6+ following WMMAs).
//  - A fragments: 2-slot ping-pong via unroll-by-2 body (constant indices,
//    each refill covered by the other slot's 8 WMMAs).
// acc[rb][nc] : rb in {0,1} (rows m0+16*rb), nc in {0..3} (cols n0+16*nc).
// ---------------------------------------------------------------------------
__device__ __forceinline__ void gemm_32x64(const bf16_t* __restrict__ X,
                                           const bf16_t* __restrict__ W,
                                           int m0, int n0, int lane,
                                           v8f acc[2][4]) {
  const bf16_t* x0 = X + (size_t)m0 * CDIM;
  const bf16_t* x1 = X + (size_t)(m0 + 16) * CDIM;
  const bf16_t* w0 = W + (size_t)n0 * CDIM;

  v16bf a0[2], a1[2], b[4];
  a0[0] = load_frag_A(x0, CDIM, lane);            // chunk 0
  a1[0] = load_frag_A(x1, CDIM, lane);
  a0[1] = load_frag_A(x0 + 32, CDIM, lane);       // chunk 1
  a1[1] = load_frag_A(x1 + 32, CDIM, lane);
  b[0] = load_frag_B(w0 + 0 * 16 * CDIM, CDIM, lane);
  b[1] = load_frag_B(w0 + 1 * 16 * CDIM, CDIM, lane);
  b[2] = load_frag_B(w0 + 2 * 16 * CDIM, CDIM, lane);
  b[3] = load_frag_B(w0 + 3 * 16 * CDIM, CDIM, lane);

#define MMA_PAIR(s, nc)                                                      \
  acc[0][nc] = wmma_bf16(a0[s], b[nc], acc[0][nc]);                          \
  acc[1][nc] = wmma_bf16(a1[s], b[nc], acc[1][nc]);

// consume current B chunk with A slot s, refilling each b[nc] (-> k offset kk)
// right after its last consumer.
#define MMA_HALF_REFILL(s, kk)                                               \
  MMA_PAIR(s, 0) b[0] = load_frag_B(w0 + 0 * 16 * CDIM + (kk), CDIM, lane);  \
  MMA_PAIR(s, 1) b[1] = load_frag_B(w0 + 1 * 16 * CDIM + (kk), CDIM, lane);  \
  MMA_PAIR(s, 2) b[2] = load_frag_B(w0 + 2 * 16 * CDIM + (kk), CDIM, lane);  \
  MMA_PAIR(s, 3) b[3] = load_frag_B(w0 + 3 * 16 * CDIM + (kk), CDIM, lane);

  // Invariant entering iteration k0: slot0 = chunk k0-64, slot1 = chunk k0-32,
  // b[] = chunk k0-64.
#pragma unroll 1
  for (int k0 = 64; k0 < CDIM; k0 += 64) {
    MMA_HALF_REFILL(0, k0 - 32)
    a0[0] = load_frag_A(x0 + k0, CDIM, lane);
    a1[0] = load_frag_A(x1 + k0, CDIM, lane);
    MMA_HALF_REFILL(1, k0)
    a0[1] = load_frag_A(x0 + k0 + 32, CDIM, lane);   // k0+32 <= CDIM-32
    a1[1] = load_frag_A(x1 + k0 + 32, CDIM, lane);
  }
  // drain: slot0 = chunk CDIM-64 (b[] = CDIM-64), slot1 = chunk CDIM-32
  MMA_HALF_REFILL(0, CDIM - 32)
  MMA_PAIR(1, 0)
  MMA_PAIR(1, 1)
  MMA_PAIR(1, 2)
  MMA_PAIR(1, 3)

#undef MMA_PAIR
#undef MMA_HALF_REFILL
}

// ---------------------------------------------------------------------------
// QKV projection: y = x @ W^T + b.  grid = (BT/256, C/64, 3), block = 256.
// z=0 -> K  [B,H,T,D] bf16 ; z=1 -> Q [B,H,T,D] bf16 ; z=2 -> V^T [B,H,D,T] bf16
// ---------------------------------------------------------------------------
__global__ __launch_bounds__(256) void qkv_gemm(
    const bf16_t* __restrict__ xb,
    const bf16_t* __restrict__ wkb, const bf16_t* __restrict__ wqb,
    const bf16_t* __restrict__ wvb,
    const float* __restrict__ bk, const float* __restrict__ bq,
    const float* __restrict__ bv,
    bf16_t* __restrict__ ko, bf16_t* __restrict__ qo, bf16_t* __restrict__ vo) {
  const int lane = threadIdx.x & 31;
  const int wave = threadIdx.x >> 5;
  const int m0 = blockIdx.x * 256 + wave * 32;     // row tile (token index)
  const int n0 = blockIdx.y * 64;                  // col tile (= head blockIdx.y)
  const int z  = blockIdx.z;
  const bf16_t* W    = (z == 0) ? wkb : (z == 1) ? wqb : wvb;
  const float*  bias = (z == 0) ? bk  : (z == 1) ? bq  : bv;

  const v8f zero = {0.f,0.f,0.f,0.f,0.f,0.f,0.f,0.f};
  v8f acc[2][4] = {{zero, zero, zero, zero}, {zero, zero, zero, zero}};
  gemm_32x64(xb, W, m0, n0, lane, acc);

  const int col = lane & 15, hh = lane >> 4;
  const int bidx = m0 / TSEQ;
  const int h    = blockIdx.y;                     // one 64-col tile == one head
#pragma unroll
  for (int rb = 0; rb < 2; ++rb) {
    const int t0 = (m0 + rb * 16) % TSEQ;
#pragma unroll
    for (int nc = 0; nc < 4; ++nc) {
      const int d = nc * 16 + col;
      const float bv_ = bias[n0 + d];
#pragma unroll
      for (int j = 0; j < 8; ++j) {
        const int t = t0 + j + hh * 8;
        const float y = acc[rb][nc][j] + bv_;
        if (z == 0)
          ko[(((size_t)bidx * NHEAD + h) * TSEQ + t) * HDIM + d] = (bf16_t)y;
        else if (z == 1)
          qo[(((size_t)bidx * NHEAD + h) * TSEQ + t) * HDIM + d] = (bf16_t)y;
        else
          vo[(((size_t)bidx * NHEAD + h) * HDIM + d) * TSEQ + t] = (bf16_t)y;
      }
    }
  }
}

// ---------------------------------------------------------------------------
// Flash attention.  grid = (T/128, B*H), block = 256 (8 waves x 16 q-rows).
// Writes softmax(QK^T/sqrt(D)) @ V as bf16 in [B, T, H*D] row-major.
// ---------------------------------------------------------------------------
__global__ __launch_bounds__(256) void attn_kernel(
    const bf16_t* __restrict__ qbp, const bf16_t* __restrict__ kbp,
    const bf16_t* __restrict__ vtp, bf16_t* __restrict__ ob) {
  __shared__ bf16_t Ks[32 * 72];          // 32 keys x 64 d, rows padded to 72
  __shared__ bf16_t Vs[64 * 40];          // 64 d x 32 keys, rows padded to 40
  __shared__ bf16_t Ps[8 * 16 * 40];      // per-wave P tile 16x32, padded

  const int lane = threadIdx.x & 31;
  const int wave = threadIdx.x >> 5;
  const int bh = blockIdx.y;              // b*NHEAD + h
  const int b = bh >> 4, h = bh & 15;
  const int Q0  = blockIdx.x * 128;
  const int qr0 = Q0 + wave * 16;

  // Q fragments for this wave's 16 rows (d-chunks of 32)
  const bf16_t* Qbase = qbp + ((size_t)bh * TSEQ + qr0) * HDIM;
  v16bf qa0 = load_frag_A(Qbase + 0,  HDIM, lane);
  v16bf qa1 = load_frag_A(Qbase + 32, HDIM, lane);

  const v8f zero = {0.f,0.f,0.f,0.f,0.f,0.f,0.f,0.f};
  v8f oacc[4] = {zero, zero, zero, zero};
  float mrow[8], lrow[8];
#pragma unroll
  for (int j = 0; j < 8; ++j) { mrow[j] = -1e30f; lrow[j] = 0.f; }

  const int col = lane & 15, hh = lane >> 4;
  const int smax = Q0 + 128;              // causal: keys needed up to last q row

  for (int s0 = 0; s0 < smax; s0 += 32) {
    // ---- cooperative stage of K tile (contiguous 4KB) and V^T tile into LDS
    {
      const int i = threadIdx.x;
      const uint4* ksrc = (const uint4*)(kbp + ((size_t)bh * TSEQ + s0) * HDIM);
      const int kr = i >> 3, kc = i & 7;
      *(uint4*)(&Ks[kr * 72 + kc * 8]) = ksrc[i];
      const int vr = i >> 2, vc = i & 3;
      *(uint4*)(&Vs[vr * 40 + vc * 8]) =
          *(const uint4*)(vtp + ((size_t)bh * HDIM + vr) * TSEQ + s0 + vc * 8);
      if (s0 + 32 < smax) {   // prefetch next K tile -> global_prefetch_b8
        const char* nk =
            (const char*)(kbp + ((size_t)bh * TSEQ + s0 + 32) * HDIM);
        __builtin_prefetch(nk + i * 16, 0, 1);
      }
    }
    __syncthreads();

    // ---- S = (Q K^T) * 1/sqrt(D), two 16x16 tiles covering 32 keys
    v8f sacc0 = zero, sacc1 = zero;
    sacc0 = wmma_bf16(qa0, load_frag_B(&Ks[0 * 72 + 0],  72, lane), sacc0);
    sacc0 = wmma_bf16(qa1, load_frag_B(&Ks[0 * 72 + 32], 72, lane), sacc0);
    sacc1 = wmma_bf16(qa0, load_frag_B(&Ks[16 * 72 + 0],  72, lane), sacc1);
    sacc1 = wmma_bf16(qa1, load_frag_B(&Ks[16 * 72 + 32], 72, lane), sacc1);

    // ---- causal mask (branch-free: keeps EXEC all-ones for WMMA)
    float sv0[8], sv1[8];
#pragma unroll
    for (int j = 0; j < 8; ++j) {
      const int r = qr0 + j + hh * 8;
      const float a0 = sacc0[j] * 0.125f, a1 = sacc1[j] * 0.125f;
      sv0[j] = ((s0 + col)      <= r) ? a0 : -1e30f;
      sv1[j] = ((s0 + 16 + col) <= r) ? a1 : -1e30f;
    }

    // ---- online softmax: row max / rescale / P / row sum
    float alpha[8];
    bf16_t* Pw = &Ps[wave * 16 * 40];
#pragma unroll
    for (int j = 0; j < 8; ++j) {
      float tmax = fmaxf(sv0[j], sv1[j]);
#pragma unroll
      for (int off = 8; off >= 1; off >>= 1)
        tmax = fmaxf(tmax, __shfl_xor(tmax, off, 16));
      const float mnew = fmaxf(mrow[j], tmax);
      alpha[j] = __expf(mrow[j] - mnew);
      mrow[j] = mnew;
      const float p0 = __expf(sv0[j] - mnew);
      const float p1 = __expf(sv1[j] - mnew);
      const int r = j + hh * 8;
      Pw[r * 40 + col]      = (bf16_t)p0;
      Pw[r * 40 + 16 + col] = (bf16_t)p1;
      float rs = p0 + p1;
#pragma unroll
      for (int off = 8; off >= 1; off >>= 1)
        rs += __shfl_xor(rs, off, 16);
      lrow[j] = lrow[j] * alpha[j] + rs;
    }
    asm volatile("s_wait_dscnt 0" ::: "memory");   // wave-local P store->load

#pragma unroll
    for (int nc = 0; nc < 4; ++nc)
#pragma unroll
      for (int j = 0; j < 8; ++j) oacc[nc][j] *= alpha[j];

    // ---- O += P @ V  (A = P 16x32 over keys, B columns = V^T rows)
    v16bf pa = load_frag_A(Pw, 40, lane);
#pragma unroll
    for (int nc = 0; nc < 4; ++nc) {
      v16bf vb = load_frag_B(&Vs[(nc * 16) * 40], 40, lane);
      oacc[nc] = wmma_bf16(pa, vb, oacc[nc]);
    }
    __syncthreads();
  }

  // ---- normalize + store bf16 [B, T, H*D]
#pragma unroll
  for (int j = 0; j < 8; ++j) {
    const float inv = 1.0f / lrow[j];
    const int t = qr0 + j + hh * 8;
#pragma unroll
    for (int nc = 0; nc < 4; ++nc) {
      const int d = nc * 16 + col;
      ob[((size_t)(b * TSEQ + t)) * CDIM + h * HDIM + d] =
          (bf16_t)(oacc[nc][j] * inv);
    }
  }
}

// ---------------------------------------------------------------------------
// Output projection: out = y @ Wp^T + bp, fp32 output.  grid = (BT/256, C/64)
// ---------------------------------------------------------------------------
__global__ __launch_bounds__(256) void proj_gemm(
    const bf16_t* __restrict__ yb, const bf16_t* __restrict__ wpb,
    const float* __restrict__ bp, float* __restrict__ out) {
  const int lane = threadIdx.x & 31;
  const int wave = threadIdx.x >> 5;
  const int m0 = blockIdx.x * 256 + wave * 32;
  const int n0 = blockIdx.y * 64;

  const v8f zero = {0.f,0.f,0.f,0.f,0.f,0.f,0.f,0.f};
  v8f acc[2][4] = {{zero, zero, zero, zero}, {zero, zero, zero, zero}};
  gemm_32x64(yb, wpb, m0, n0, lane, acc);

  const int col = lane & 15, hh = lane >> 4;
#pragma unroll
  for (int rb = 0; rb < 2; ++rb) {
#pragma unroll
    for (int nc = 0; nc < 4; ++nc) {
      const int c = n0 + nc * 16 + col;
      const float bias = bp[c];
#pragma unroll
      for (int j = 0; j < 8; ++j) {
        const int r = m0 + rb * 16 + j + hh * 8;
        out[(size_t)r * CDIM + c] = acc[rb][nc][j] + bias;
      }
    }
  }
}

// ---------------------------------------------------------------------------
extern "C" void kernel_launch(void* const* d_in, const int* in_sizes, int n_in,
                              void* d_out, int out_size, void* d_ws,
                              size_t ws_size, hipStream_t stream) {
  const float* x  = (const float*)d_in[0];
  const float* Wk = (const float*)d_in[1];
  const float* bk = (const float*)d_in[2];
  const float* Wq = (const float*)d_in[3];
  const float* bq = (const float*)d_in[4];
  const float* Wv = (const float*)d_in[5];
  const float* bv = (const float*)d_in[6];
  const float* Wp = (const float*)d_in[7];
  const float* bp = (const float*)d_in[8];
  (void)in_sizes; (void)n_in; (void)out_size; (void)ws_size;

  const size_t NX = (size_t)4 * TSEQ * CDIM;   // 8,388,608 tokens*feat
  const size_t NW = (size_t)CDIM * CDIM;       // 1,048,576

  char* ws = (char*)d_ws;
  bf16_t* xb  = (bf16_t*)(ws);                                   // 16 MiB
  bf16_t* wkb = (bf16_t*)(ws + 16u * 1024 * 1024);
  bf16_t* wqb = wkb + NW;
  bf16_t* wvb = wqb + NW;
  bf16_t* wpb = wvb + NW;                                        // +8 MiB
  bf16_t* kb  = (bf16_t*)(ws + 24u * 1024 * 1024);               // 16 MiB
  bf16_t* qb  = kb + NX;                                         // 16 MiB
  bf16_t* vtb = qb + NX;                                         // 16 MiB
  bf16_t* ob  = vtb + NX;                                        // 16 MiB

  cvt_f32_bf16_v4<<<2048, 256, 0, stream>>>((const float4*)x,
                                            (ushort4*)xb, (int)(NX / 4));
  cvt_f32_bf16_v4<<<256, 256, 0, stream>>>((const float4*)Wk,
                                           (ushort4*)wkb, (int)(NW / 4));
  cvt_f32_bf16_v4<<<256, 256, 0, stream>>>((const float4*)Wq,
                                           (ushort4*)wqb, (int)(NW / 4));
  cvt_f32_bf16_v4<<<256, 256, 0, stream>>>((const float4*)Wv,
                                           (ushort4*)wvb, (int)(NW / 4));
  cvt_f32_bf16_v4<<<256, 256, 0, stream>>>((const float4*)Wp,
                                           (ushort4*)wpb, (int)(NW / 4));

  dim3 gq(32, 16, 3);
  qkv_gemm<<<gq, 256, 0, stream>>>(xb, wkb, wqb, wvb, bk, bq, bv, kb, qb, vtb);

  dim3 ga(TSEQ / 128, 4 * NHEAD);
  attn_kernel<<<ga, 256, 0, stream>>>(qb, kb, vtb, ob);

  dim3 gp(32, 16);
  proj_gemm<<<gp, 256, 0, stream>>>(ob, wpb, bp, (float*)d_out);
}
